// QuantLinear_8143257993936
// MI455X (gfx1250) — compile-verified, matrix operand order checked by reading, not run on hardware
//
#include <hip/hip_runtime.h>

// ---------------- problem constants (match reference) ----------------
#define IN_F   4096
#define OUT_F  4096
#define GS     128
#define NG     (IN_F / GS)        // 32 groups
// ---------------- tiling ----------------
#define BM     128                // rows of X per workgroup
#define BN     128                // out features per workgroup
#define BK     128                // k-chunk == group size (one scale per chunk)
#define THREADS 256               // 8 wave32 waves
#define NWAVE   8

typedef __attribute__((ext_vector_type(16))) __bf16 v16bf;
typedef __attribute__((ext_vector_type(8)))  __bf16 v8bf;
typedef __attribute__((ext_vector_type(8)))  float  v8f;
typedef __attribute__((ext_vector_type(4)))  int    v4i;

#define XV 16   // async b128 copies per thread: 128*128 fp32 / (256 thr * 4 fl) = 16
#define WV 8    // int4 loads per thread: 128*64 int32 / (256 thr * 4) = 8

// ---- CDNA5 async global->LDS copy (ASYNCcnt-tracked DMA path) ----
__device__ __forceinline__ void async_copy_b128(const float* g, float* l) {
#if __has_builtin(__builtin_amdgcn_global_load_async_to_lds_b128)
    __builtin_amdgcn_global_load_async_to_lds_b128(
        (__attribute__((address_space(1))) v4i*)(void*)g,
        (__attribute__((address_space(3))) v4i*)(void*)l,
        0 /*imm offset*/, 0 /*cpol*/);
#else
    unsigned laddr = (unsigned)(unsigned long long)
        (__attribute__((address_space(3))) char*)(void*)l;
    asm volatile("global_load_async_to_lds_b128 %0, %1, off"
                 :: "v"(laddr), "v"(g) : "memory");
#endif
}

__device__ __forceinline__ void wait_async0() {
#if __has_builtin(__builtin_amdgcn_s_wait_asynccnt)
    __builtin_amdgcn_s_wait_asynccnt(0);
#else
    asm volatile("s_wait_asynccnt 0x0" ::: "memory");
#endif
}

__global__ __launch_bounds__(THREADS, 1)
void q4_gemm_wmma_bf16(const float* __restrict__ x,
                       const int*   __restrict__ qw,
                       const float* __restrict__ scale,
                       float*       __restrict__ out)
{
    // LDS: X raw fp32 double-buffer (128KB) + W bf16 double-buffer (64KB) = 192KB.
    // Indexed DIRECTLY (integer stage offsets) so clang keeps addrspace(3) -> ds ops.
    __shared__ __align__(16) float          xsh[2 * BM * BK];
    __shared__ __align__(16) unsigned short bsh[2 * BN * BK];

    const int tid  = threadIdx.x;
    const int wave = tid >> 5;
    const int lane = tid & 31;
    const int n0   = blockIdx.x * BN;
    const int m0   = blockIdx.y * BM;

    const int laneM = lane & 15;        // row/col within 16x16 tile
    const int kHalf = (lane >> 4) * 8;  // which K half this lane holds

    v8f acc[8];
    const v8f vzero = {};
    #pragma unroll
    for (int t = 0; t < 8; ++t) acc[t] = vzero;

    int4  wr[WV];
    float sr[WV];

    // ---- X tile: async DMA global -> LDS (raw fp32, no VGPR staging) ----
    auto async_x = [&](int kc, int stage) {
        const int so = stage * (BM * BK);
        #pragma unroll
        for (int it = 0; it < XV; ++it) {
            int idx4 = it * THREADS + tid;   // float4 index in 128x128 tile
            int r = idx4 >> 5;               // 32 float4 per row
            int c = idx4 & 31;
            async_copy_b128(&x[(size_t)(m0 + r) * IN_F + (size_t)kc * BK + c * 4],
                            &xsh[so + r * BK + c * 4]);
        }
    };

    // ---- W tile: global -> registers (packed int4 + one scale per row) ----
    auto load_w = [&](int kc) {
        #pragma unroll
        for (int it = 0; it < WV; ++it) {
            int idx4 = it * THREADS + tid;   // int4 index; 16 int4 per row
            int r = idx4 >> 4;
            int c = idx4 & 15;
            wr[it] = *reinterpret_cast<const int4*>(
                &qw[(size_t)(n0 + r) * (IN_F / 2) + (size_t)kc * (BK / 2) + c * 4]);
            sr[it] = scale[(size_t)(n0 + r) * NG + kc];
        }
    };

    // ---- dequant registers -> bf16 LDS stage ----
    auto stage_w = [&](int stage) {
        const int so = stage * (BN * BK);
        #pragma unroll
        for (int it = 0; it < WV; ++it) {
            int idx4 = it * THREADS + tid;
            int r = idx4 >> 4;
            int c = idx4 & 15;
            const float s = sr[it];
            const int q0 = wr[it].x, q1 = wr[it].y, q2 = wr[it].z, q3 = wr[it].w;
            v8bf v;
            // even index = high nibble, odd index = low nibble, offset-8, times scale
            v[0] = (__bf16)((float)(((q0 >> 4) & 15) - 8) * s);
            v[1] = (__bf16)((float)((q0 & 15) - 8) * s);
            v[2] = (__bf16)((float)(((q1 >> 4) & 15) - 8) * s);
            v[3] = (__bf16)((float)((q1 & 15) - 8) * s);
            v[4] = (__bf16)((float)(((q2 >> 4) & 15) - 8) * s);
            v[5] = (__bf16)((float)((q2 & 15) - 8) * s);
            v[6] = (__bf16)((float)(((q3 >> 4) & 15) - 8) * s);
            v[7] = (__bf16)((float)((q3 & 15) - 8) * s);
            *reinterpret_cast<v8bf*>(&bsh[so + r * BK + c * 8]) = v;   // ds_store_b128
        }
    };

    // ---- WMMA over one stage: K = 128 = 4 x K32 steps ----
    // Per step: build A from fp32 LDS (cvt->bf16); B fragments preloaded in
    // groups of 4 tiles so 4 WMMAs run per LDS wait with bounded VGPR use.
    auto compute = [&](int stage) {
        const int sx = stage * (BM * BK);
        const int sb = stage * (BN * BK);
        const int mrow = wave * 16 + laneM;
        #pragma unroll
        for (int ks = 0; ks < 4; ++ks) {
            const int kb = ks * 32 + kHalf;
            // A fragment: 16 fp32 (ds_load_b128 x4) -> 16 bf16 (v_cvt_pk_bf16_f32)
            float4 a0 = *reinterpret_cast<const float4*>(&xsh[sx + mrow * BK + kb]);
            float4 a1 = *reinterpret_cast<const float4*>(&xsh[sx + mrow * BK + kb + 4]);
            float4 a2 = *reinterpret_cast<const float4*>(&xsh[sx + mrow * BK + kb + 16]);
            float4 a3 = *reinterpret_cast<const float4*>(&xsh[sx + mrow * BK + kb + 20]);
            union { v16bf v; v8bf h[2]; } A;
            A.v[0]  = (__bf16)a0.x; A.v[1]  = (__bf16)a0.y;
            A.v[2]  = (__bf16)a0.z; A.v[3]  = (__bf16)a0.w;
            A.v[4]  = (__bf16)a1.x; A.v[5]  = (__bf16)a1.y;
            A.v[6]  = (__bf16)a1.z; A.v[7]  = (__bf16)a1.w;
            A.v[8]  = (__bf16)a2.x; A.v[9]  = (__bf16)a2.y;
            A.v[10] = (__bf16)a2.z; A.v[11] = (__bf16)a2.w;
            A.v[12] = (__bf16)a3.x; A.v[13] = (__bf16)a3.y;
            A.v[14] = (__bf16)a3.z; A.v[15] = (__bf16)a3.w;
            #pragma unroll
            for (int g = 0; g < 2; ++g) {            // two groups of 4 tiles
                v8bf blo[4], bhi[4];
                #pragma unroll
                for (int t = 0; t < 4; ++t) {
                    const int nrow = (g * 4 + t) * 16 + laneM;
                    blo[t] = *reinterpret_cast<const v8bf*>(&bsh[sb + nrow * BK + kb]);
                    bhi[t] = *reinterpret_cast<const v8bf*>(&bsh[sb + nrow * BK + kb + 16]);
                }
                #pragma unroll
                for (int t = 0; t < 4; ++t) {
                    union { v16bf v; v8bf h[2]; } Bv;
                    Bv.h[0] = blo[t];
                    Bv.h[1] = bhi[t];
                    acc[g * 4 + t] = __builtin_amdgcn_wmma_f32_16x16x32_bf16(
                        false, A.v, false, Bv.v, (short)0, acc[g * 4 + t], false, false);
                }
            }
        }
    };

    // ---------------- pipelined main loop over 32 k-chunks ----------------
    async_x(0, 0);
    load_w(0);
    stage_w(0);
    wait_async0();
    __syncthreads();

    for (int kc = 0; kc < NG; ++kc) {
        const int cur = kc & 1;
        if (kc + 1 < NG) {
            async_x(kc + 1, cur ^ 1);   // DMA engine fills next X buffer
            load_w(kc + 1);             // weight loads in flight under WMMAs
        }
        compute(cur);
        if (kc + 1 < NG) stage_w(cur ^ 1);
        wait_async0();                  // own async writes done before barrier
        __syncthreads();
    }

    // ---------------- write C: 16x16 f32 D layout ----------------
    // VGPR r holds M=r (lanes 0-15) / M=r+8 (lanes 16-31), lane%16 = N
    const int rbase = m0 + wave * 16 + ((lane >> 4) << 3);
    #pragma unroll
    for (int t = 0; t < 8; ++t) {
        const int col = n0 + t * 16 + laneM;
        #pragma unroll
        for (int r = 0; r < 8; ++r) {
            out[(size_t)(rbase + r) * OUT_F + col] = acc[t][r];
        }
    }
}

extern "C" void kernel_launch(void* const* d_in, const int* in_sizes, int n_in,
                              void* d_out, int out_size, void* d_ws, size_t ws_size,
                              hipStream_t stream) {
    const float* x     = (const float*)d_in[0];
    const int*   qw    = (const int*)d_in[1];
    const float* scale = (const float*)d_in[2];
    float*       out   = (float*)d_out;

    const int rows = in_sizes[0] / IN_F;             // batch = 4096
    dim3 grid(OUT_F / BN, rows / BM);                // 32 x 32 workgroups
    q4_gemm_wmma_bf16<<<grid, THREADS, 0, stream>>>(x, qw, scale, out);
}
